// Attention_1795296330192
// MI455X (gfx1250) — matrix operand order repaired and checked by reading, not compiled, not run on hardware
//
#include <hip/hip_runtime.h>
#include <hip/hip_bf16.h>

// ---------------------------------------------------------------------------
// Problem constants (from reference): B=8, N=1024, C=768, H=12, D=64
// ---------------------------------------------------------------------------
#define BD 8
#define ND 1024
#define CD 768
#define HD 12
#define DD 64
#define C3 2304                 // 3*C
#define MROWS (BD * ND)         // 8192
#define SCALEF 0.125f
#define NEGBIG (-1.0e30f)

typedef __bf16 bf16;
typedef __bf16 v16bf __attribute__((ext_vector_type(16)));
typedef float  v8f   __attribute__((ext_vector_type(8)));

// Workspace layout (bytes). Total ~68.7 MB.
#define OFF_XB     ((size_t)0)                         // x bf16      [8192,768]
#define OFF_WQKVT  (OFF_XB    + (size_t)MROWS*CD*2)    // qkv_w^T bf16 [768,2304]
#define OFF_WPROJT (OFF_WQKVT + (size_t)CD*C3*2)       // proj_w^T bf16 [768,768]
#define OFF_Q      (OFF_WPROJT+ (size_t)CD*CD*2)       // Q bf16 [B,H,N,D]
#define OFF_KT     (OFF_Q     + (size_t)BD*HD*ND*DD*2) // K^T bf16 [B,H,D,N]
#define OFF_V      (OFF_KT    + (size_t)BD*HD*ND*DD*2) // V bf16 [B,H,N,D]
#define OFF_CTX    (OFF_V     + (size_t)BD*HD*ND*DD*2) // ctx bf16 [B,N,C]
#define OFF_MBITS  (OFF_CTX   + (size_t)MROWS*CD*2)    // mask bits [B,N,N/8]

// ---------------------------------------------------------------------------
// WMMA + data-movement helpers
// ---------------------------------------------------------------------------
__device__ __forceinline__ v8f wmma_bf16(v16bf a, v16bf b, v8f c) {
    return __builtin_amdgcn_wmma_f32_16x16x32_bf16(
        /*neg_a=*/false, a, /*neg_b=*/false, b,
        /*c_mod=*/(short)0, c, /*reuse_a=*/false, /*reuse_b=*/false);
}

// A-operand (16x32, 16-bit) per ISA layout:
//   lanes 0-15 : VGPR0-3 hold K=0..7  pairs, VGPR4-7 hold K=16..23
//   lanes 16-31: VGPR0-3 hold K=8..15 pairs, VGPR4-7 hold K=24..31
__device__ __forceinline__ v16bf load_a(const bf16* rowptr, int half) {
    union { v16bf v; uint4 q[2]; } u;
    u.q[0] = *(const uint4*)(rowptr + half * 8);
    u.q[1] = *(const uint4*)(rowptr + 16 + half * 8);
    return u.v;
}

// B-operand (32x16, 16-bit): lane l holds column n=l%16, K in same order as A.
__device__ __forceinline__ v16bf load_b(const bf16* colptr, int ld, int half) {
    v16bf v;
#pragma unroll
    for (int s = 0; s < 8; ++s) {
        v[s]     = colptr[(8 * half + s) * ld];
        v[8 + s] = colptr[(16 + 8 * half + s) * ld];
    }
    return v;
}

__device__ __forceinline__ v8f v8f_zero() {
    v8f z = {0.f, 0.f, 0.f, 0.f, 0.f, 0.f, 0.f, 0.f};
    return z;
}

// Async global->LDS copy of 16 bytes per lane (GLOBAL_LOAD_ASYNC_TO_LDS_B128,
// tracked by ASYNCcnt). lds_byte_addr = low 32 bits of the flat shared pointer
// (addrspacecast local->flat is {aperture_hi, lds_offset} per ISA aperture rules).
__device__ __forceinline__ void async_copy16(unsigned lds_byte_addr, const bf16* gsrc) {
    asm volatile("global_load_async_to_lds_b128 %0, %1, off"
                 :: "v"(lds_byte_addr), "v"((unsigned long long)(size_t)gsrc)
                 : "memory");
}
__device__ __forceinline__ void wait_async0() {
    asm volatile("s_wait_asynccnt 0x0" ::: "memory");
}
__device__ __forceinline__ unsigned lds_addr(const void* p) {
    return (unsigned)(size_t)p;
}

// ---------------------------------------------------------------------------
// Pre-pass kernels
// ---------------------------------------------------------------------------
__global__ void cast_f32_bf16(const float* __restrict__ in, bf16* __restrict__ out, int n) {
    int i = blockIdx.x * blockDim.x + threadIdx.x;
    if (i < n) out[i] = (bf16)in[i];
}

// wt[k*R + r] = (bf16) w[r*Ck + k]   (writes coalesced)
__global__ void transpose_cast(const float* __restrict__ w, bf16* __restrict__ wt,
                               int R, int Ck) {
    int i = blockIdx.x * blockDim.x + threadIdx.x;
    if (i < R * Ck) {
        int k = i / R, r = i - k * R;
        wt[i] = (bf16)w[(size_t)r * Ck + k];
    }
}

// Pack int32 0/1 mask into bits: byte i covers elements 8i..8i+7 (bit j = elem 8i+j)
__global__ void pack_mask(const int* __restrict__ m, unsigned char* __restrict__ mb, int nbytes) {
    int i = blockIdx.x * blockDim.x + threadIdx.x;
    if (i < nbytes) {
        const int* p = m + (size_t)i * 8;
        unsigned v = 0;
#pragma unroll
        for (int j = 0; j < 8; ++j) v |= (unsigned)(p[j] != 0) << j;
        mb[i] = (unsigned char)v;
    }
}

// ---------------------------------------------------------------------------
// QKV GEMM: out[m, j] = sum_k xb[m,k] * wqkvt[k,j] + bias[j]
// B tile (32x64, shared by all 8 waves) staged to LDS with double-buffered
// async copies. grid: (C3/64, MROWS/128), block 256.
// ---------------------------------------------------------------------------
__global__ void qkv_gemm(const bf16* __restrict__ xb, const bf16* __restrict__ wt,
                         const float* __restrict__ bias,
                         bf16* __restrict__ Q, bf16* __restrict__ Kt, bf16* __restrict__ V) {
    __shared__ __align__(16) bf16 bstage[2][32][72];   // rows padded 128B->144B

    const int tid = threadIdx.x;
    const int lane = tid & 31, wave = tid >> 5;
    const int half = lane >> 4, lm = lane & 15;
    const int m0 = blockIdx.y * 128 + wave * 16;
    const int n0 = blockIdx.x * 64;
    const bf16* arow = xb + (size_t)(m0 + lm) * CD;

    // Staging map: thread -> (row k = tid/8, 16B chunk = tid%8) of the 32x64 tile
    const int srow = tid >> 3, scol = (tid & 7) * 8;
    const bf16* gsrc = wt + (size_t)srow * C3 + n0 + scol;   // + k0*C3 per stage
    const unsigned ldst0 = lds_addr(&bstage[0][srow][scol]);
    const unsigned ldst1 = lds_addr(&bstage[1][srow][scol]);

    async_copy16(ldst0, gsrc);                                // prologue: k0 = 0

    v8f acc[4];
#pragma unroll
    for (int t = 0; t < 4; ++t) acc[t] = v8f_zero();

    for (int k0 = 0; k0 < CD; k0 += 32) {
        const int cur = (k0 >> 5) & 1;
        wait_async0();
        __syncthreads();                                      // bstage[cur] visible to all
        if (k0 + 32 < CD)
            async_copy16(cur ? ldst0 : ldst1, gsrc + (size_t)(k0 + 32) * C3);

        v16bf a = load_a(arow + k0, half);
        const bf16* bl = &bstage[cur][0][lm];
#pragma unroll
        for (int t = 0; t < 4; ++t)
            acc[t] = wmma_bf16(a, load_b(bl + 16 * t, 72, half), acc[t]);
    }

#pragma unroll
    for (int t = 0; t < 4; ++t) {
        const int col = n0 + 16 * t + lm;
        const float bv = bias[col];
        const int which = col / CD;      // 0=q, 1=k, 2=v
        const int cc = col % CD;
        const int h = cc / DD, d = cc % DD;
#pragma unroll
        for (int r = 0; r < 8; ++r) {
            const int mr = m0 + r + 8 * half;        // C-layout: M = r + 8*half
            const int bidx = mr >> 10, nidx = mr & (ND - 1);
            const bf16 o = (bf16)(acc[t][r] + bv);
            const size_t bh = (size_t)(bidx * HD + h);
            if (which == 0)      Q[(bh * ND + nidx) * DD + d]  = o;
            else if (which == 1) Kt[(bh * DD + d) * ND + nidx] = o;
            else                 V[(bh * ND + nidx) * DD + d]  = o;
        }
    }
}

// ---------------------------------------------------------------------------
// Flash attention: grid (N/128, B*H), block 256 (8 waves x 16 q-rows).
// K^T and V tiles (shared by all waves) staged via double-buffered async
// copies; keys streamed in steps of 32 with online softmax.
// ---------------------------------------------------------------------------
__global__ void flash_attn(const bf16* __restrict__ Q, const bf16* __restrict__ Kt,
                           const bf16* __restrict__ V, const unsigned char* __restrict__ mbits,
                           bf16* __restrict__ ctx) {
    __shared__ __align__(16) bf16 kbuf[2][64][40];    // K^T tile [d][key], 64B->80B rows
    __shared__ __align__(16) bf16 vbuf[2][32][72];    // V tile [key][d], 128B->144B rows
    __shared__ __align__(16) bf16 plds[8][16 * 40];   // per-wave P bounce, padded rows

    const int tid = threadIdx.x;
    const int lane = tid & 31, wave = tid >> 5;
    const int half = lane >> 4, lm = lane & 15;
    const int bh = blockIdx.y, b = bh / HD, h = bh % HD;
    const int qm0 = blockIdx.x * 128 + wave * 16;

    // Q tile as two A-operands (d=0..31, d=32..63)
    const bf16* qbase = Q + ((size_t)bh * ND + qm0 + lm) * DD;
    const v16bf aq0 = load_a(qbase, half);
    const v16bf aq1 = load_a(qbase + 32, half);

    const bf16* ktbase = Kt + (size_t)bh * DD * ND;   // [d, n], n contiguous
    const bf16* vbase  = V  + (size_t)bh * ND * DD;   // [n, d], d contiguous
    const unsigned char* mrow = mbits + ((size_t)b * ND + qm0) * (ND / 8);

    // Staging maps (one b128 each for K-tile and V-tile per thread per step)
    const int kd = tid >> 2, kc = (tid & 3) * 8;      // K: row d, chunk of 32 keys
    const int vk = tid >> 3, vc = (tid & 7) * 8;      // V: row key, chunk of 64 d
    const bf16* kg = ktbase + (size_t)kd * ND + kc;   // + k0 per stage
    const bf16* vg = vbase + (size_t)vk * DD + vc;    // + k0*DD per stage
    const unsigned kl0 = lds_addr(&kbuf[0][kd][kc]), kl1 = lds_addr(&kbuf[1][kd][kc]);
    const unsigned vl0 = lds_addr(&vbuf[0][vk][vc]), vl1 = lds_addr(&vbuf[1][vk][vc]);

    async_copy16(kl0, kg);                            // prologue: k0 = 0
    async_copy16(vl0, vg);

    float mrv[8], lrv[8];
    v8f o[4];
#pragma unroll
    for (int r = 0; r < 8; ++r) { mrv[r] = NEGBIG; lrv[r] = 0.f; }
#pragma unroll
    for (int t = 0; t < 4; ++t) o[t] = v8f_zero();

    for (int k0 = 0; k0 < ND; k0 += 32) {
        const int cur = (k0 >> 5) & 1;
        wait_async0();
        __syncthreads();                              // tiles for this step visible
        if (k0 + 32 < ND) {
            async_copy16(cur ? kl0 : kl1, kg + (k0 + 32));
            async_copy16(cur ? vl0 : vl1, vg + (size_t)(k0 + 32) * DD);
        }

        // ---- S = Q * K^T for 32 keys (two 16-col C tiles) ----
        v8f s0 = v8f_zero(), s1 = v8f_zero();
        {
            v16bf b00 = load_b(&kbuf[cur][0][lm], 40, half);        // keys +0..15,  d 0..31
            v16bf b01 = load_b(&kbuf[cur][0][16 + lm], 40, half);   // keys +16..31, d 0..31
            v16bf b10 = load_b(&kbuf[cur][32][lm], 40, half);       // keys +0..15,  d 32..63
            v16bf b11 = load_b(&kbuf[cur][32][16 + lm], 40, half);
            s0 = wmma_bf16(aq0, b00, s0);
            s0 = wmma_bf16(aq1, b10, s0);
            s1 = wmma_bf16(aq0, b01, s1);
            s1 = wmma_bf16(aq1, b11, s1);
        }

        // ---- mask + scale ----
        const int key0 = k0 + lm, key1 = k0 + 16 + lm;
#pragma unroll
        for (int r = 0; r < 8; ++r) {
            const unsigned char* mb = mrow + (size_t)(r + 8 * half) * (ND / 8);
            s0[r] = ((mb[key0 >> 3] >> (key0 & 7)) & 1) ? s0[r] * SCALEF : NEGBIG;
            s1[r] = ((mb[key1 >> 3] >> (key1 & 7)) & 1) ? s1[r] * SCALEF : NEGBIG;
        }

        // ---- online softmax update (row r+8*half lives across 16 lanes) ----
#pragma unroll
        for (int r = 0; r < 8; ++r) {
            float v = fmaxf(s0[r], s1[r]);
            v = fmaxf(v, __shfl_xor(v, 1, 32));
            v = fmaxf(v, __shfl_xor(v, 2, 32));
            v = fmaxf(v, __shfl_xor(v, 4, 32));
            v = fmaxf(v, __shfl_xor(v, 8, 32));
            const float mnew = fmaxf(mrv[r], v);
            const float fac = __expf(mrv[r] - mnew);
            mrv[r] = mnew;
            s0[r] = __expf(s0[r] - mnew);
            s1[r] = __expf(s1[r] - mnew);
            float sum = s0[r] + s1[r];
            sum += __shfl_xor(sum, 1, 32);
            sum += __shfl_xor(sum, 2, 32);
            sum += __shfl_xor(sum, 4, 32);
            sum += __shfl_xor(sum, 8, 32);
            lrv[r] = lrv[r] * fac + sum;
#pragma unroll
            for (int t = 0; t < 4; ++t) o[t][r] *= fac;
        }

        // ---- P: C-layout f32 -> LDS -> A-layout bf16 (per-wave region) ----
        bf16* pl = &plds[wave][0];
#pragma unroll
        for (int r = 0; r < 8; ++r) {
            const int rr = r + 8 * half;
            pl[rr * 40 + lm]      = (bf16)s0[r];
            pl[rr * 40 + 16 + lm] = (bf16)s1[r];
        }
        asm volatile("s_wait_dscnt 0" ::: "memory");  // DS in-order per wave; wait data
        const v16bf pa = load_a(pl + lm * 40, half);

        // ---- O += P * V (K-dim = 32 keys, 4 d-tiles) ----
#pragma unroll
        for (int t2 = 0; t2 < 4; ++t2) {
            v16bf bv = load_b(&vbuf[cur][0][16 * t2 + lm], 72, half);
            o[t2] = wmma_bf16(pa, bv, o[t2]);
        }
    }

    // ---- normalize + store context in [b, n, h*64+d] ----
#pragma unroll
    for (int r = 0; r < 8; ++r) {
        const float inv = lrv[r] > 0.f ? 1.f / lrv[r] : 0.f;
        const int nidx = qm0 + r + 8 * half;
        bf16* crow = ctx + ((size_t)b * ND + nidx) * CD + h * DD;
#pragma unroll
        for (int t2 = 0; t2 < 4; ++t2)
            crow[16 * t2 + lm] = (bf16)(o[t2][r] * inv);
    }
}

// ---------------------------------------------------------------------------
// Output projection: out[m,j] = sum_k ctx[m,k] * wprojt[k,j] + bias[j]  (f32 out)
// Same async-staged structure as qkv_gemm. grid: (CD/64, MROWS/128), block 256.
// ---------------------------------------------------------------------------
__global__ void proj_gemm(const bf16* __restrict__ ctx, const bf16* __restrict__ wt,
                          const float* __restrict__ bias, float* __restrict__ out) {
    __shared__ __align__(16) bf16 bstage[2][32][72];

    const int tid = threadIdx.x;
    const int lane = tid & 31, wave = tid >> 5;
    const int half = lane >> 4, lm = lane & 15;
    const int m0 = blockIdx.y * 128 + wave * 16;
    const int n0 = blockIdx.x * 64;
    const bf16* arow = ctx + (size_t)(m0 + lm) * CD;

    const int srow = tid >> 3, scol = (tid & 7) * 8;
    const bf16* gsrc = wt + (size_t)srow * CD + n0 + scol;
    const unsigned ldst0 = lds_addr(&bstage[0][srow][scol]);
    const unsigned ldst1 = lds_addr(&bstage[1][srow][scol]);

    async_copy16(ldst0, gsrc);

    v8f acc[4];
#pragma unroll
    for (int t = 0; t < 4; ++t) acc[t] = v8f_zero();

    for (int k0 = 0; k0 < CD; k0 += 32) {
        const int cur = (k0 >> 5) & 1;
        wait_async0();
        __syncthreads();
        if (k0 + 32 < CD)
            async_copy16(cur ? ldst0 : ldst1, gsrc + (size_t)(k0 + 32) * CD);

        v16bf a = load_a(arow + k0, half);
        const bf16* bl = &bstage[cur][0][lm];
#pragma unroll
        for (int t = 0; t < 4; ++t)
            acc[t] = wmma_bf16(a, load_b(bl + 16 * t, 72, half), acc[t]);
    }

#pragma unroll
    for (int t = 0; t < 4; ++t) {
        const int col = n0 + 16 * t + lm;
        const float bv = bias[col];
#pragma unroll
        for (int r = 0; r < 8; ++r) {
            const int mr = m0 + r + 8 * half;
            out[(size_t)mr * CD + col] = acc[t][r] + bv;
        }
    }
}

// ---------------------------------------------------------------------------
// Launch
// ---------------------------------------------------------------------------
extern "C" void kernel_launch(void* const* d_in, const int* in_sizes, int n_in,
                              void* d_out, int out_size, void* d_ws, size_t ws_size,
                              hipStream_t stream) {
    const float* x      = (const float*)d_in[0];
    const int*   mask   = (const int*)d_in[1];
    const float* qkv_w  = (const float*)d_in[2];
    const float* qkv_b  = (const float*)d_in[3];
    const float* proj_w = (const float*)d_in[4];
    const float* proj_b = (const float*)d_in[5];
    float* out = (float*)d_out;

    char* ws = (char*)d_ws;
    bf16* xb     = (bf16*)(ws + OFF_XB);
    bf16* wqkvt  = (bf16*)(ws + OFF_WQKVT);
    bf16* wprojt = (bf16*)(ws + OFF_WPROJT);
    bf16* Qb     = (bf16*)(ws + OFF_Q);
    bf16* Ktb    = (bf16*)(ws + OFF_KT);
    bf16* Vb     = (bf16*)(ws + OFF_V);
    bf16* ctx    = (bf16*)(ws + OFF_CTX);
    unsigned char* mbits = (unsigned char*)(ws + OFF_MBITS);

    // Pre-passes
    {
        int n = MROWS * CD;
        cast_f32_bf16<<<(n + 255) / 256, 256, 0, stream>>>(x, xb, n);
    }
    {
        int n = C3 * CD;
        transpose_cast<<<(n + 255) / 256, 256, 0, stream>>>(qkv_w, wqkvt, C3, CD);
    }
    {
        int n = CD * CD;
        transpose_cast<<<(n + 255) / 256, 256, 0, stream>>>(proj_w, wprojt, CD, CD);
    }
    {
        int n = BD * ND * ND / 8;
        pack_mask<<<(n + 255) / 256, 256, 0, stream>>>(mask, mbits, n);
    }

    // QKV projection -> Q / K^T / V (bf16)
    qkv_gemm<<<dim3(C3 / 64, MROWS / 128), 256, 0, stream>>>(xb, wqkvt, qkv_b, Qb, Ktb, Vb);

    // Flash attention -> context [B,N,C] bf16
    flash_attn<<<dim3(ND / 128, BD * HD), 256, 0, stream>>>(Qb, Ktb, Vb, mbits, ctx);

    // Output projection -> f32 out
    proj_gemm<<<dim3(CD / 64, MROWS / 128), 256, 0, stream>>>(ctx, wprojt, proj_b, out);
}
